// CausalSelfAttention_24592982736943
// MI455X (gfx1250) — compile-verified, matrix operand order checked by reading, not run on hardware
//
#include <hip/hip_runtime.h>
#include <hip/hip_bf16.h>

#define D_MODEL 1024
#define N_HEADS 16
#define HEAD_DIM 64
#define TSEQ 2048
#define BATCH 4

typedef __bf16 bf16;
typedef __attribute__((ext_vector_type(16))) __bf16 v16bf;
typedef __attribute__((ext_vector_type(8)))  float  v8f;

// ---------------- CDNA5 async global->LDS copy (guarded) ----------------
#if defined(__has_builtin)
#if __has_builtin(__builtin_amdgcn_global_load_async_to_lds_b128)
#define HAVE_ASYNC_LDS 1
#endif
#endif

#if defined(HAVE_ASYNC_LDS)
// Builtin signature (from toolchain diagnostic): (v4i AS1*, v4i AS3*, imm, imm)
typedef int v4i_b128 __attribute__((vector_size(16)));
typedef __attribute__((address_space(1))) v4i_b128* as1_v4i;
typedef __attribute__((address_space(3))) v4i_b128* as3_v4i;

__device__ __forceinline__ void cp_async16(bf16* l, const bf16* g) {
  // AS1 pointer = full 64-bit VA; AS3 pointer = low 32 bits (LDS offset,
  // per ISA flat->LDS mapping which discards upper bits).
  __builtin_amdgcn_global_load_async_to_lds_b128(
      (as1_v4i)(unsigned long long)(uintptr_t)g,
      (as3_v4i)(unsigned int)(uintptr_t)l, 0, 0);
}
__device__ __forceinline__ void wait_async() {
#if __has_builtin(__builtin_amdgcn_s_wait_asynccnt)
  __builtin_amdgcn_s_wait_asynccnt(0);
#else
  asm volatile("s_wait_asynccnt 0x0" ::: "memory");
#endif
}
#else
__device__ __forceinline__ void cp_async16(bf16* l, const bf16* g) {
  *(uint4*)l = *(const uint4*)g;
}
__device__ __forceinline__ void wait_async() {}
#endif

union FragU { uint4 u[2]; v16bf v; };

// Load one 16x32 bf16 WMMA fragment (A or B operand) from a K-contiguous
// row-major buffer. Layout per CDNA5 ISA 7.12.2:
// lane L: row = row0 + (L%16); chunks K = {koff..+7, koff+16..+23},
// koff = (L>=16)?8:0, all offset by k0.
__device__ __forceinline__ v16bf load_frag_ld(const bf16* base, int row0, int ld,
                                              int k0) {
  const int lane = threadIdx.x & 31;
  const bf16* p = base + (size_t)(row0 + (lane & 15)) * ld + k0 + ((lane >> 4) << 3);
  FragU f;
  f.u[0] = *(const uint4*)(p);
  f.u[1] = *(const uint4*)(p + 16);
  return f.v;
}

__global__ void cast_f32_bf16(const float* __restrict__ in, bf16* __restrict__ out,
                              long long n) {
  long long i = (long long)blockIdx.x * blockDim.x + threadIdx.x;
  const long long stride = (long long)gridDim.x * blockDim.x;
  for (; i < n; i += stride) out[i] = (bf16)in[i];
}

// C[M x NN] (fp32) = A[M x K] (bf16, row-major) * W[NN x K]^T (bf16, row-major).
// Block = 256 threads = 8 waves; tile 128x128; K-step 32; double-buffered LDS
// with async global->LDS staging overlapped with WMMA on the other buffer.
template <int NN>
__global__ __launch_bounds__(256) void gemm_bf16(const bf16* __restrict__ A,
                                                 const bf16* __restrict__ W,
                                                 float* __restrict__ C,
                                                 int M, int K) {
  __shared__ alignas(16) bf16 sA[2][128 * 32];
  __shared__ alignas(16) bf16 sB[2][128 * 32];
  const int bm = blockIdx.y * 128;
  const int bn = blockIdx.x * 128;
  const int tid  = threadIdx.x;
  const int wave = tid >> 5;
  const int lane = tid & 31;
  const int wm = (wave >> 1) * 32;  // wave's M offset within tile
  const int wn = (wave & 1) * 64;   // wave's N offset within tile

  v8f acc[2][4];
  #pragma unroll
  for (int i = 0; i < 2; i++)
    #pragma unroll
    for (int j = 0; j < 4; j++) acc[i][j] = v8f{};

  const int lrow = tid >> 1;        // 0..127
  const int lcol = (tid & 1) * 16;  // 0 or 16 (bf16 elements)
  const int loff = lrow * 32 + lcol;
  const bf16* gA = A + (size_t)(bm + lrow) * K + lcol;
  const bf16* gW = W + (size_t)(bn + lrow) * K + lcol;

  auto stage = [&](int buf, int k0) {
    cp_async16(&sA[buf][loff],     gA + k0);
    cp_async16(&sA[buf][loff + 8], gA + k0 + 8);
    cp_async16(&sB[buf][loff],     gW + k0);
    cp_async16(&sB[buf][loff + 8], gW + k0 + 8);
  };
  auto compute = [&](int buf) {
    v16bf aF[2], bF[4];
    #pragma unroll
    for (int i = 0; i < 2; i++) aF[i] = load_frag_ld(&sA[buf][0], wm + i * 16, 32, 0);
    #pragma unroll
    for (int j = 0; j < 4; j++) bF[j] = load_frag_ld(&sB[buf][0], wn + j * 16, 32, 0);
    #pragma unroll
    for (int i = 0; i < 2; i++)
      #pragma unroll
      for (int j = 0; j < 4; j++)
        acc[i][j] = __builtin_amdgcn_wmma_f32_16x16x32_bf16(
            false, aF[i], false, bF[j], (short)0, acc[i][j], false, false);
  };

  stage(0, 0);
  wait_async();
  __syncthreads();
  int ping = 0;
  for (int k0 = 32; k0 < K; k0 += 32) {
    stage(ping ^ 1, k0);                    // prefetch next tile (async engine)
    __builtin_prefetch(gA + k0 + 32, 0, 0); // hint the tile after that into GL2
    compute(ping);                          // WMMA on current tile
    wait_async();
    __syncthreads();
    ping ^= 1;
  }
  compute(ping);

  // Epilogue: NN is compile-time -> row strides fold into store immediates.
  const int rhi = (lane >> 4) * 8;
  const int cn  = lane & 15;
  #pragma unroll
  for (int i = 0; i < 2; i++)
    #pragma unroll
    for (int j = 0; j < 4; j++) {
      float* cp = C + (size_t)(bm + wm + i * 16 + rhi) * NN + (bn + wn + j * 16 + cn);
      #pragma unroll
      for (int r = 0; r < 8; r++) cp[r * NN] = acc[i][j][r];
    }
}

// qkv fp32 [B,T,3*D_MODEL] -> RoPE'd q,k and raw v as bf16 in [b,h,t,d] layout.
// One thread handles rotation pair (i, i+32) within one head.
__global__ void rope_split(const float* __restrict__ qkv, bf16* __restrict__ q,
                           bf16* __restrict__ k, bf16* __restrict__ v) {
  int idx = blockIdx.x * blockDim.x + threadIdx.x;  // B*H*T*32 threads
  const int i = idx & 31;            idx >>= 5;
  const int t = idx & (TSEQ - 1);    idx >>= 11;
  const int h = idx & (N_HEADS - 1);
  const int b = idx >> 4;
  const float* src = qkv + (size_t)(b * TSEQ + t) * (3 * D_MODEL) + h * HEAD_DIM;

  const float inv = __powf(10000.0f, -(float)(2 * i) * (1.0f / 64.0f));
  float s, c;
  __sincosf((float)t * inv, &s, &c);

  const size_t dst = ((size_t)(b * N_HEADS + h) * TSEQ + t) * HEAD_DIM + i;
  const float q1 = src[i], q2 = src[i + 32];
  q[dst]      = (bf16)(q1 * c - q2 * s);
  q[dst + 32] = (bf16)(q2 * c + q1 * s);
  const float k1 = src[D_MODEL + i], k2 = src[D_MODEL + i + 32];
  k[dst]      = (bf16)(k1 * c - k2 * s);
  k[dst + 32] = (bf16)(k2 * c + k1 * s);
  v[dst]      = (bf16)src[2 * D_MODEL + i];
  v[dst + 32] = (bf16)src[2 * D_MODEL + i + 32];
}

// Flash attention: block = (qt, b*h), 128 threads = 4 waves.
// Wave w owns query rows [qt*64 + w*16, +16). Streams 64-key tiles kt<=qt.
// Output written as bf16 [b,t,c] for the final projection GEMM.
__global__ __launch_bounds__(128) void flash_attn(const bf16* __restrict__ Qg,
                                                  const bf16* __restrict__ Kg,
                                                  const bf16* __restrict__ Vg,
                                                  bf16* __restrict__ Og) {
  __shared__ alignas(16) bf16 sK[64 * 64];       // [key][d]
  __shared__ alignas(16) bf16 sVT[64 * 64];      // [d][key] (transposed)
  __shared__ alignas(16) bf16 sP[4][16 * 64];    // per-wave P staging [row][key]

  const int qt = blockIdx.x;   // 0..31
  const int bh = blockIdx.y;   // 0..63
  const int b = bh >> 4, h = bh & 15;
  const int tid = threadIdx.x, wave = tid >> 5, lane = tid & 31;
  const float scale = 0.125f;  // 1/sqrt(HEAD_DIM)

  // Resident Q fragments: 16 rows x 64 K -> two 16x32 fragments.
  const bf16* Qbase = Qg + ((size_t)bh * TSEQ + qt * 64 + wave * 16) * HEAD_DIM;
  v16bf qF[2];
  qF[0] = load_frag_ld(Qbase, 0, HEAD_DIM, 0);
  qF[1] = load_frag_ld(Qbase, 0, HEAD_DIM, 32);

  float m[8], l[8];
  v8f O[4];
  #pragma unroll
  for (int r = 0; r < 8; r++) { m[r] = -1e30f; l[r] = 0.0f; }
  #pragma unroll
  for (int j = 0; j < 4; j++) O[j] = v8f{};

  const int rhi = (lane >> 4) * 8;
  const int cn  = lane & 15;
  const int qrow_tile = wave * 16 + rhi;  // +r = local query row in 64-tile

  const int lrow  = tid >> 1;         // 0..63 (loader row)
  const int lcoff = (tid & 1) * 32;   // 0 or 32 elements

  for (int kt = 0; kt <= qt; kt++) {
    const bf16* Kt = Kg + ((size_t)bh * TSEQ + kt * 64) * HEAD_DIM;
    const bf16* Vt = Vg + ((size_t)bh * TSEQ + kt * 64) * HEAD_DIM;
    // K tile: [key][d], async straight copy into LDS.
    #pragma unroll
    for (int c = 0; c < 4; c++)
      cp_async16(sK + lrow * 64 + lcoff + c * 8, Kt + lrow * 64 + lcoff + c * 8);
    // V tile: store transposed so B-fragment reads are contiguous.
    #pragma unroll
    for (int c = 0; c < 4; c++) {
      uint4 raw = *(const uint4*)(Vt + lrow * 64 + lcoff + c * 8);
      const bf16* e = (const bf16*)&raw;
      #pragma unroll
      for (int x = 0; x < 8; x++) sVT[(lcoff + c * 8 + x) * 64 + lrow] = e[x];
    }
    wait_async();
    __syncthreads();

    // S = Q K^T : 16x64 per wave = 4 n-subtiles, 2 K-steps each.
    v8f S[4];
    #pragma unroll
    for (int j = 0; j < 4; j++) {
      v16bf b0 = load_frag_ld(sK, j * 16, 64, 0);
      v16bf b1 = load_frag_ld(sK, j * 16, 64, 32);
      v8f s = v8f{};
      s = __builtin_amdgcn_wmma_f32_16x16x32_bf16(false, qF[0], false, b0,
                                                  (short)0, s, false, false);
      s = __builtin_amdgcn_wmma_f32_16x16x32_bf16(false, qF[1], false, b1,
                                                  (short)0, s, false, false);
      S[j] = s;
    }

    // scale + causal mask (only the diagonal tile needs masking)
    #pragma unroll
    for (int j = 0; j < 4; j++)
      #pragma unroll
      for (int r = 0; r < 8; r++) {
        float sv = S[j][r] * scale;
        if (kt == qt && (j * 16 + cn) > (qrow_tile + r)) sv = -1e30f;
        S[j][r] = sv;
      }

    // Online softmax. Each row lives in one 16-lane half at vector index r.
    float alpha[8];
    #pragma unroll
    for (int r = 0; r < 8; r++) {
      float v0 = fmaxf(fmaxf(S[0][r], S[1][r]), fmaxf(S[2][r], S[3][r]));
      #pragma unroll
      for (int off = 8; off >= 1; off >>= 1) v0 = fmaxf(v0, __shfl_xor(v0, off, 32));
      const float mn = fmaxf(m[r], v0);
      alpha[r] = __expf(m[r] - mn);
      m[r] = mn;
    }
    float rsum[8];
    #pragma unroll
    for (int r = 0; r < 8; r++) rsum[r] = 0.0f;
    #pragma unroll
    for (int j = 0; j < 4; j++)
      #pragma unroll
      for (int r = 0; r < 8; r++) {
        const float p = __expf(S[j][r] - m[r]);
        rsum[r] += p;
        sP[wave][(r + rhi) * 64 + j * 16 + cn] = (bf16)p;  // C-layout -> LDS
      }
    #pragma unroll
    for (int r = 0; r < 8; r++) {
      float t0 = rsum[r];
      #pragma unroll
      for (int off = 8; off >= 1; off >>= 1) t0 += __shfl_xor(t0, off, 32);
      l[r] = l[r] * alpha[r] + t0;
    }
    #pragma unroll
    for (int j = 0; j < 4; j++)
      #pragma unroll
      for (int r = 0; r < 8; r++) O[j][r] *= alpha[r];

    // O += P V (P re-read from LDS in A-fragment layout; V from transposed LDS)
    v16bf pF0 = load_frag_ld(sP[wave], 0, 64, 0);
    v16bf pF1 = load_frag_ld(sP[wave], 0, 64, 32);
    #pragma unroll
    for (int j = 0; j < 4; j++) {
      v16bf v0 = load_frag_ld(sVT, j * 16, 64, 0);
      v16bf v1 = load_frag_ld(sVT, j * 16, 64, 32);
      O[j] = __builtin_amdgcn_wmma_f32_16x16x32_bf16(false, pF0, false, v0,
                                                     (short)0, O[j], false, false);
      O[j] = __builtin_amdgcn_wmma_f32_16x16x32_bf16(false, pF1, false, v1,
                                                     (short)0, O[j], false, false);
    }
    __syncthreads();
  }

  // Epilogue: normalize and write bf16 [b][t][h*64+d]
  #pragma unroll
  for (int j = 0; j < 4; j++)
    #pragma unroll
    for (int r = 0; r < 8; r++) {
      const int t = qt * 64 + qrow_tile + r;
      const size_t o = (size_t)(b * TSEQ + t) * D_MODEL + h * HEAD_DIM + j * 16 + cn;
      Og[o] = (bf16)(O[j][r] / l[r]);
    }
}

extern "C" void kernel_launch(void* const* d_in, const int* in_sizes, int n_in,
                              void* d_out, int out_size, void* d_ws, size_t ws_size,
                              hipStream_t stream) {
  const float* x     = (const float*)d_in[0];
  const float* w_qkv = (const float*)d_in[1];
  const float* w_out = (const float*)d_in[2];
  float* out = (float*)d_out;

  const size_t M = (size_t)BATCH * TSEQ;  // 8192 tokens
  char* ws = (char*)d_ws;
  size_t off = 0;
  auto carve = [&](size_t bytes) -> void* {
    void* p = ws + off;
    off += (bytes + 255) & ~(size_t)255;
    return p;
  };
  bf16*  x_bf    = (bf16*)carve(M * D_MODEL * 2);
  bf16*  wqkv_bf = (bf16*)carve((size_t)3 * D_MODEL * D_MODEL * 2);
  bf16*  wout_bf = (bf16*)carve((size_t)D_MODEL * D_MODEL * 2);
  float* qkv     = (float*)carve(M * 3 * D_MODEL * 4);
  bf16*  q_bf    = (bf16*)carve(M * D_MODEL * 2);
  bf16*  k_bf    = (bf16*)carve(M * D_MODEL * 2);
  bf16*  v_bf    = (bf16*)carve(M * D_MODEL * 2);
  bf16*  a_bf    = (bf16*)carve(M * D_MODEL * 2);

  cast_f32_bf16<<<2048, 256, 0, stream>>>(x, x_bf, (long long)(M * D_MODEL));
  cast_f32_bf16<<<1024, 256, 0, stream>>>(w_qkv, wqkv_bf,
                                          (long long)3 * D_MODEL * D_MODEL);
  cast_f32_bf16<<<512, 256, 0, stream>>>(w_out, wout_bf,
                                         (long long)D_MODEL * D_MODEL);

  dim3 g1(3 * D_MODEL / 128, (unsigned)(M / 128));
  gemm_bf16<3 * D_MODEL><<<g1, 256, 0, stream>>>(x_bf, wqkv_bf, qkv, (int)M, D_MODEL);

  const int rope_threads = BATCH * N_HEADS * TSEQ * 32;
  rope_split<<<rope_threads / 256, 256, 0, stream>>>(qkv, q_bf, k_bf, v_bf);

  dim3 g2(TSEQ / 64, BATCH * N_HEADS);
  flash_attn<<<g2, 128, 0, stream>>>(q_bf, k_bf, v_bf, a_bf);

  dim3 g3(D_MODEL / 128, (unsigned)(M / 128));
  gemm_bf16<D_MODEL><<<g3, 256, 0, stream>>>(a_bf, wout_bf, out, (int)M, D_MODEL);
}
